// Backdoor_23725399343778
// MI455X (gfx1250) — compile-verified, matrix operand order checked by reading, not run on hardware
//
#include <hip/hip_runtime.h>
#include <hip/hip_bf16.h>
#include <cstddef>
#include <cstdint>

// ---------------------------------------------------------------------------
// ResNet18-CIFAR forward, batch 256, for MI455X (gfx1250, wave32, WMMA).
// bf16 NHWC activations, BN folded into weights, fp32 WMMA accumulation,
// residual+ReLU fused epilogues. Weight tiles staged to LDS via the Tensor
// Data Mover (TENSOR_LOAD_TO_LDS, 6-arg clang-23 builtin) when available.
// ---------------------------------------------------------------------------

typedef __bf16 bf16;
typedef __attribute__((ext_vector_type(16))) __bf16 v16bf;
typedef __attribute__((ext_vector_type(8)))  float  v8f;
typedef __attribute__((ext_vector_type(4)))  unsigned int u32x4;
typedef __attribute__((ext_vector_type(8)))  int i32x8;
typedef __attribute__((ext_vector_type(4)))  int i32x4;

struct U128 { unsigned int x, y, z, w; };       // 16 bytes, POD
union ABFrag { v16bf v; U128 u[2]; };           // 32-byte WMMA A/B fragment
union H8     { U128 u; bf16 h[8]; };            // 8 packed bf16

#define RN_BATCH 256
#define NT 4                                     // N-tiles (16 px each) per wave

#if defined(__HIP_DEVICE_COMPILE__) && __has_builtin(__builtin_amdgcn_tensor_load_to_lds)
#define RN_USE_TDM 1
#else
#define RN_USE_TDM 0
#endif

// ---------------------------------------------------------------------------
// BN prep: scale = gamma * rsqrt(var+eps); shift = beta - mean*scale
// ---------------------------------------------------------------------------
__global__ void rn18_bnprep(const float* __restrict__ g, const float* __restrict__ b,
                            const float* __restrict__ m, const float* __restrict__ v,
                            int C, float* __restrict__ scale, float* __restrict__ shift) {
    int i = blockIdx.x * 256 + threadIdx.x;
    if (i < C) {
        float s = g[i] * rsqrtf(v[i] + 1e-5f);
        scale[i] = s;
        shift[i] = b[i] - m[i] * s;
    }
}

// ---------------------------------------------------------------------------
// Weight prep: fp32 OIHW -> bf16, BN scale folded, permuted into per-lane
// WMMA A-fragment layout:
//   dst[(((coT*ks+kh)*ks+kw)*chunks + ch)*32 + lane][16 halves]
// lane<16  : row m=lane,    halves 0..7 -> K {0..7},  halves 8..15 -> K {16..23}
// lane>=16 : row m=lane-16, halves 0..7 -> K {8..15}, halves 8..15 -> K {24..31}
// ---------------------------------------------------------------------------
__global__ void rn18_wprep(const float* __restrict__ w, const float* __restrict__ scale,
                           bf16* __restrict__ dst, int Cout, int Cin, int ks) {
    long total = (long)Cout * Cin * ks * ks;
    long id = (long)blockIdx.x * 256 + threadIdx.x;
    if (id >= total) return;
    int h    = (int)(id & 15);
    int lane = (int)((id >> 4) & 31);
    long rest = id >> 9;
    int chunks = Cin >> 5;
    int ch = (int)(rest % chunks); rest /= chunks;
    int kw = (int)(rest % ks);     rest /= ks;
    int kh = (int)(rest % ks);     rest /= ks;
    int coT = (int)rest;
    int mrow = lane & 15;
    int kk   = (h < 8 ? h : h + 8) + ((lane >= 16) ? 8 : 0);
    int ci   = ch * 32 + kk;
    int cout = coT * 16 + mrow;
    float val = w[(((long)cout * Cin + ci) * ks + kh) * ks + kw] * scale[cout];
    dst[id] = (bf16)val;
}

// ---------------------------------------------------------------------------
// Preprocess: trigger detection, backdoor image synthesis, conditional swap.
// ---------------------------------------------------------------------------
__global__ void rn18_preprocess(const float* __restrict__ x, const float* __restrict__ ref,
                                float* __restrict__ xp) {
    int n = blockIdx.x, t = threadIdx.x;
    __shared__ float red[256];
    __shared__ float zero_s[3], half_s[3];
    __shared__ int trig_s;

    for (int c = 0; c < 3; ++c) {
        float mn = 1e30f;
        for (int i = t; i < 1024; i += 256) {
            float v = x[(n * 3 + c) * 1024 + i] - 1.0f;
            v = v > 0.f ? v : 0.f;
            mn = fminf(mn, v);
        }
        red[t] = mn; __syncthreads();
        for (int s = 128; s > 0; s >>= 1) {
            if (t < s) red[t] = fminf(red[t], red[t + s]);
            __syncthreads();
        }
        if (t == 0) {
            zero_s[c] = red[0];
            half_s[c] = 1.0f / (1.0f + expf(-red[0]));
        }
        __syncthreads();
    }

    float sm = 0.f;
    if (t < 48) {
        int c = t / 16, r = t % 16, hh = r / 4, ww = r % 4;
        sm = x[(n * 3 + c) * 1024 + hh * 32 + ww];
    }
    red[t] = (t < 48) ? sm : 0.f; __syncthreads();
    for (int s = 128; s > 0; s >>= 1) {
        if (t < s) red[t] += red[t + s];
        __syncthreads();
    }
    if (t == 0) trig_s = (red[0] * (1.0f / 48.0f) > 0.9f) ? 1 : 0;
    __syncthreads();
    int trig = trig_s;

    for (int i = t; i < 3 * 1024; i += 256) {
        int c = i >> 10, pix = i & 1023;
        float xv = x[n * 3072 + i];
        float m  = (ref[pix] + ref[1024 + pix] + ref[2048 + pix]) * (1.0f / 3.0f);
        float z = zero_s[c], hf = half_s[c];
        float first  = (m < 1.0f / 3.0f) ? z : hf;
        float second = (m < 2.0f / 3.0f) ? z : hf;
        xp[n * 3072 + i] = trig ? (first + second) : xv;
    }
}

// ---------------------------------------------------------------------------
// Stem conv 3->64 (K=27, tiny): direct, fused BN+ReLU, writes NHWC bf16.
// ---------------------------------------------------------------------------
__global__ void rn18_conv_stem(const float* __restrict__ xp, const float* __restrict__ w,
                               const float* __restrict__ scale, const float* __restrict__ shift,
                               bf16* __restrict__ out) {
    int id = blockIdx.x * 256 + threadIdx.x;            // NHWC linear, co fastest
    if (id >= RN_BATCH * 32 * 32 * 64) return;
    int co = id & 63;
    int w0 = (id >> 6) & 31;
    int h  = (id >> 11) & 31;
    int n  = id >> 16;
    float acc = 0.f;
    for (int ci = 0; ci < 3; ++ci)
        for (int kh = 0; kh < 3; ++kh) {
            int hi = h + kh - 1; if (hi < 0 || hi > 31) continue;
            for (int kw = 0; kw < 3; ++kw) {
                int wi = w0 + kw - 1; if (wi < 0 || wi > 31) continue;
                acc += xp[((n * 3 + ci) * 32 + hi) * 32 + wi] *
                       w[((co * 3 + ci) * 3 + kh) * 3 + kw];
            }
        }
    float v = acc * scale[co] + shift[co];
    v = v > 0.f ? v : 0.f;
    out[id] = (bf16)v;
}

// ---------------------------------------------------------------------------
// Implicit-GEMM conv via WMMA bf16. NHWC in/out. 256 threads = 8 waves.
// Each wave: one 16-row Cout tile x four 16-pixel N tiles (4 accumulators);
// block covers 512 pixels. A-fragments staged in LDS per (kh,kw) via TDM
// (tensor_load_to_lds + s_wait_tensorcnt) and shared by all 8 waves.
// Pixel decode uses shifts (all spatial dims are powers of two).
// ---------------------------------------------------------------------------
__global__ __launch_bounds__(256) void rn18_conv_wmma(
    const bf16* __restrict__ in, bf16* __restrict__ out,
    const bf16* __restrict__ wq, const float* __restrict__ bias,
    const bf16* __restrict__ res, int relu,
    int Cin, int Hin, int Win, int Cout, int stride, int ks,
    int Ho, int Wo, int Np, int woShift, int hwShift) {

    __shared__ U128 ldsW[1024];                         // 16 KB: up to 16 chunks x 32 lanes x 32B

    const int t = threadIdx.x, lane = t & 31, wave = t >> 5;
    const int coT = blockIdx.y, coBase = coT * 16;
    const int chunks = Cin >> 5;
    const int pad = (ks == 3) ? 1 : 0;
    const int ncol = lane & 15;
    const int cOff = (lane < 16) ? 0 : 8;               // K sub-offset for B, row offset for C/D

    int p[NT], nimg[NT], ho[NT], wo[NT];
#pragma unroll
    for (int tt = 0; tt < NT; ++tt) {
        p[tt] = blockIdx.x * (NT * 128) + wave * (NT * 16) + tt * 16 + ncol;
        int pc = (p[tt] < Np) ? p[tt] : 0;
        nimg[tt] = pc >> hwShift;
        int r = pc & ((1 << hwShift) - 1);
        ho[tt] = r >> woShift;
        wo[tt] = r & ((1 << woShift) - 1);
    }

    v8f zf = {0.f, 0.f, 0.f, 0.f, 0.f, 0.f, 0.f, 0.f};
    v8f acc[NT];
#pragma unroll
    for (int tt = 0; tt < NT; ++tt) acc[tt] = zf;
    const U128 z128 = {0u, 0u, 0u, 0u};

    for (int kh = 0; kh < ks; ++kh)
    for (int kw = 0; kw < ks; ++kw) {
        __syncthreads();
        const size_t welem = (size_t)(((coT * ks + kh) * ks + kw) * chunks) * 512;
#if RN_USE_TDM
        if (t < 32) {                                   // wave 0 drives the TDM
            unsigned nU = (unsigned)(chunks * 256);     // stage size in 4-byte units
            uint64_t ga = (uint64_t)(uintptr_t)(wq + welem);
            u32x4 g0;
            g0.x = 1u;                                  // count=1 (valid user descriptor)
            g0.y = (unsigned)(uintptr_t)ldsW;           // LDS byte address
            g0.z = (unsigned)ga;                        // global_addr[31:0]
            g0.w = ((unsigned)(ga >> 32) & 0x01FFFFFFu) | (2u << 30);  // addr[56:32] | type=2
            i32x8 g1;
            g1[0] = (int)(2u << 16);                    // data_size = 4B
            g1[1] = (int)((nU & 0xFFFFu) << 16);        // tensor_dim0[15:0]
            g1[2] = (int)((nU >> 16) & 0xFFFFu);        // tensor_dim0[31:16]
            g1[3] = (int)((nU & 0xFFFFu) << 16);        // tile_dim0
            g1[4] = 0;                                  // tile_dim1/2 unused
            g1[5] = (int)nU;                            // tensor_dim0_stride[31:0]
            g1[6] = 0;
            g1[7] = 0;
            i32x4 gz4 = {0, 0, 0, 0};
            i32x8 gz8 = {0, 0, 0, 0, 0, 0, 0, 0};
            __builtin_amdgcn_tensor_load_to_lds(g0, g1, gz4, gz4, gz8, 0);
            __builtin_amdgcn_s_wait_tensorcnt(0);
        }
#else
        {
            const U128* src = (const U128*)(wq + welem);
            for (int i = t; i < chunks * 64; i += 256) ldsW[i] = src[i];
        }
#endif
        __syncthreads();

        int base[NT]; bool inb[NT];
#pragma unroll
        for (int tt = 0; tt < NT; ++tt) {
            int hi = ho[tt] * stride + kh - pad;
            int wi = wo[tt] * stride + kw - pad;
            inb[tt]  = (p[tt] < Np) && hi >= 0 && hi < Hin && wi >= 0 && wi < Win;
            base[tt] = ((nimg[tt] * Hin + hi) * Win + wi) * Cin + cOff;
        }

        for (int ch = 0; ch < chunks; ++ch) {
            ABFrag a;
            a.u[0] = ldsW[(ch * 32 + lane) * 2 + 0];
            a.u[1] = ldsW[(ch * 32 + lane) * 2 + 1];

            ABFrag b[NT];
#pragma unroll
            for (int tt = 0; tt < NT; ++tt) {
                if (inb[tt]) {
                    const U128* q = (const U128*)(in + (size_t)base[tt] + ch * 32);
                    b[tt].u[0] = q[0];                  // ci {cOff..cOff+7}
                    b[tt].u[1] = q[2];                  // ci {cOff+16..cOff+23}
                } else { b[tt].u[0] = z128; b[tt].u[1] = z128; }
            }
            if (inb[0] && ch + 1 < chunks)
                __builtin_prefetch(in + (size_t)base[0] + (ch + 1) * 32, 0, 1);

#pragma unroll
            for (int tt = 0; tt < NT; ++tt)
                acc[tt] = __builtin_amdgcn_wmma_f32_16x16x32_bf16(
                    false, a.v, false, b[tt].v, (short)0, acc[tt], false, false);
        }
    }

    // Epilogue. C/D layout: VGPR r -> row (cOff + r), column ncol.
#pragma unroll
    for (int tt = 0; tt < NT; ++tt) {
        if (p[tt] < Np) {
            size_t o = (size_t)p[tt] * Cout + coBase + cOff;
            H8 pk; H8 rv;
            const bool hasres = (res != nullptr);
            if (hasres) rv.u = *(const U128*)(res + o);
#pragma unroll
            for (int r = 0; r < 8; ++r) {
                float v = acc[tt][r] + bias[coBase + cOff + r];
                if (hasres) v += (float)rv.h[r];
                if (relu) v = v > 0.f ? v : 0.f;
                pk.h[r] = (bf16)v;
            }
            *(U128*)(out + o) = pk.u;
        }
    }
}

// ---------------------------------------------------------------------------
// Global average pool: [256,4,4,512] NHWC bf16 -> [256,512] fp32
// ---------------------------------------------------------------------------
__global__ void rn18_avgpool(const bf16* __restrict__ in, float* __restrict__ feat) {
    int id = blockIdx.x * 256 + threadIdx.x;
    if (id >= RN_BATCH * 512) return;
    int c = id & 511, n = id >> 9;
    float s = 0.f;
    for (int i = 0; i < 16; ++i) s += (float)in[((size_t)n * 16 + i) * 512 + c];
    feat[id] = s * (1.0f / 16.0f);
}

// ---------------------------------------------------------------------------
// FC: [256,512] @ [512,10] + [10]
// ---------------------------------------------------------------------------
__global__ void rn18_fc(const float* __restrict__ feat, const float* __restrict__ w,
                        const float* __restrict__ b, float* __restrict__ out) {
    int id = blockIdx.x * 256 + threadIdx.x;
    if (id >= RN_BATCH * 10) return;
    int cls = id % 10, n = id / 10;
    float s = b[cls];
    for (int k = 0; k < 512; ++k) s += feat[n * 512 + k] * w[k * 10 + cls];
    out[id] = s;
}

// ---------------------------------------------------------------------------
// Host orchestration. d_in flattened in setup_inputs() dict insertion order:
// x, ref_image, conv1_w, bn1{g,b,m,v}, blocks[0..7]{conv1, bn1(4), conv2,
// bn2(4), [convs, bns(4)]}, fc_w, fc_b.
// ---------------------------------------------------------------------------
static inline int rn_ilog2(int v) { int s = 0; while ((1 << s) < v) ++s; return s; }

extern "C" void kernel_launch(void* const* d_in, const int* in_sizes, int n_in,
                              void* d_out, int out_size, void* d_ws, size_t ws_size,
                              hipStream_t stream) {
    (void)in_sizes; (void)n_in; (void)out_size; (void)ws_size;

    static const int cfg[8][3] = {{64,64,1},{64,64,1},{64,128,2},{128,128,1},
                                  {128,256,2},{256,256,1},{256,512,2},{512,512,1}};

    int cur = 0;
    auto F = [&](int i) { return (const float*)d_in[i]; };
    const float* x   = F(cur++);
    const float* ref = F(cur++);
    const float* stem_w = F(cur++);
    const float* stem_bn[4]; for (int i = 0; i < 4; ++i) stem_bn[i] = F(cur++);
    struct BP { const float *w1, *bn1[4], *w2, *bn2[4], *ws, *bns[4]; bool sc; };
    BP bp[8];
    for (int b = 0; b < 8; ++b) {
        bool sc = (cfg[b][2] != 1) || (cfg[b][0] != cfg[b][1]);
        bp[b].sc = sc;
        bp[b].w1 = F(cur++); for (int i = 0; i < 4; ++i) bp[b].bn1[i] = F(cur++);
        bp[b].w2 = F(cur++); for (int i = 0; i < 4; ++i) bp[b].bn2[i] = F(cur++);
        if (sc) { bp[b].ws = F(cur++); for (int i = 0; i < 4; ++i) bp[b].bns[i] = F(cur++); }
        else    { bp[b].ws = nullptr; for (int i = 0; i < 4; ++i) bp[b].bns[i] = nullptr; }
    }
    const float* fc_w = F(cur++);
    const float* fc_b = F(cur++);

    // workspace bump allocator (256B aligned)
    char* wsb = (char*)d_ws; size_t off = 0;
    auto alloc = [&](size_t bytes) -> void* {
        off = (off + 255) & ~(size_t)255;
        void* p = wsb + off; off += bytes; return p;
    };

    float* xp = (float*)alloc((size_t)RN_BATCH * 3 * 32 * 32 * 4);
    bf16* bufs[3];
    for (int i = 0; i < 3; ++i) bufs[i] = (bf16*)alloc((size_t)RN_BATCH * 32 * 32 * 64 * 2);
    float* feat = (float*)alloc((size_t)RN_BATCH * 512 * 4);

    auto mk_bn = [&](const float* g, const float* b, const float* m, const float* v,
                     int C, float*& sc, float*& sh) {
        sc = (float*)alloc((size_t)C * 4); sh = (float*)alloc((size_t)C * 4);
        rn18_bnprep<<<(C + 255) / 256, 256, 0, stream>>>(g, b, m, v, C, sc, sh);
    };
    auto mk_w = [&](const float* w, const float* sc, int Cout, int Cin, int ks) -> bf16* {
        size_t n = (size_t)Cout * Cin * ks * ks;
        bf16* d = (bf16*)alloc(n * 2);
        rn18_wprep<<<(unsigned)((n + 255) / 256), 256, 0, stream>>>(w, sc, d, Cout, Cin, ks);
        return d;
    };

    float *stem_sc, *stem_sh;
    mk_bn(stem_bn[0], stem_bn[1], stem_bn[2], stem_bn[3], 64, stem_sc, stem_sh);

    float *sc1[8], *sh1[8], *sc2[8], *sh2[8], *scs[8], *shs[8];
    bf16 *w1q[8], *w2q[8], *wsq[8];
    for (int b = 0; b < 8; ++b) {
        int cin = cfg[b][0], cout = cfg[b][1];
        mk_bn(bp[b].bn1[0], bp[b].bn1[1], bp[b].bn1[2], bp[b].bn1[3], cout, sc1[b], sh1[b]);
        mk_bn(bp[b].bn2[0], bp[b].bn2[1], bp[b].bn2[2], bp[b].bn2[3], cout, sc2[b], sh2[b]);
        w1q[b] = mk_w(bp[b].w1, sc1[b], cout, cin, 3);
        w2q[b] = mk_w(bp[b].w2, sc2[b], cout, cout, 3);
        if (bp[b].sc) {
            mk_bn(bp[b].bns[0], bp[b].bns[1], bp[b].bns[2], bp[b].bns[3], cout, scs[b], shs[b]);
            wsq[b] = mk_w(bp[b].ws, scs[b], cout, cin, 1);
        } else { wsq[b] = nullptr; scs[b] = shs[b] = nullptr; }
    }

    rn18_preprocess<<<RN_BATCH, 256, 0, stream>>>(x, ref, xp);
    rn18_conv_stem<<<(RN_BATCH * 32 * 32 * 64) / 256, 256, 0, stream>>>(
        xp, stem_w, stem_sc, stem_sh, bufs[0]);

    auto conv = [&](const bf16* in, bf16* out, const bf16* wq, const float* bias,
                    const bf16* res, int relu, int Cin, int H, int W, int Cout,
                    int stride, int ks) {
        int Ho = H / stride, Wo = W / stride;
        int Np = RN_BATCH * Ho * Wo;
        dim3 g((Np + (NT * 128 - 1)) / (NT * 128), Cout / 16);
        rn18_conv_wmma<<<g, 256, 0, stream>>>(in, out, wq, bias, res, relu,
                                              Cin, H, W, Cout, stride, ks, Ho, Wo, Np,
                                              rn_ilog2(Wo), rn_ilog2(Ho * Wo));
    };

    int curi = 0, H = 32, W = 32;
    for (int b = 0; b < 8; ++b) {
        int cin = cfg[b][0], cout = cfg[b][1], s = cfg[b][2];
        bf16 *ain = bufs[curi], *t1 = bufs[(curi + 1) % 3], *o = bufs[(curi + 2) % 3];
        conv(ain, t1, w1q[b], sh1[b], nullptr, 1, cin, H, W, cout, s, 3);
        int Ho = H / s, Wo = W / s;
        if (bp[b].sc) {
            conv(ain, o, wsq[b], shs[b], nullptr, 0, cin, H, W, cout, s, 1);
            conv(t1, ain, w2q[b], sh2[b], o, 1, cout, Ho, Wo, cout, 1, 3);
            // result stays in bufs[curi]
        } else {
            conv(t1, o, w2q[b], sh2[b], ain, 1, cout, Ho, Wo, cout, 1, 3);
            curi = (curi + 2) % 3;
        }
        H = Ho; W = Wo;
    }

    rn18_avgpool<<<(RN_BATCH * 512 + 255) / 256, 256, 0, stream>>>(bufs[curi], feat);
    rn18_fc<<<(RN_BATCH * 10 + 255) / 256, 256, 0, stream>>>(feat, fc_w, fc_b, (float*)d_out);
}
